// FixedRNN_51058571215071
// MI455X (gfx1250) — matrix operand order repaired and checked by reading, not compiled, not run on hardware
//
#include <hip/hip_runtime.h>
#include <hip/hip_bf16.h>

// ---------------------------------------------------------------------------
// Ensemble encoder-decoder LSTM for MI455X (gfx1250), wave32 + WMMA bf16.
//
// One workgroup = (ensemble member e, 64-row batch tile). The whole 192-step
// recurrence runs inside the workgroup: U (128x512) lives in LDS as bf16
// (transposed, padded), h lives in LDS as bf16, c lives in f32 registers.
// Per step each of 16 waves does 32 x v_wmma_f32_16x16x32_bf16.
// ---------------------------------------------------------------------------

typedef __bf16 bf16_t;
typedef __attribute__((ext_vector_type(16))) __bf16 v16bf;
typedef __attribute__((ext_vector_type(8)))  float  v8f;

#define TDIM   192
#define CONTL  168
#define HORIZ  24
#define BATCH  4096
#define HDIM   128
#define GDIM   512      // 4*H
#define DDIM   8
#define EDIM   8
#define BT     64       // batch rows per workgroup
#define NTH    512      // 16 waves
#define UTS    136      // padded row stride (elements) for Ut  [512][136] bf16
#define HS     136      // padded row stride (elements) for h   [64][136]  bf16

// -------------------------- LDS layout (bytes) -----------------------------
#define OFF_UT 0
#define SZ_UT  (GDIM * UTS * 2)            // 139264
#define OFF_HB (OFF_UT + SZ_UT)
#define SZ_HB  (BT * HS * 2)               // 17408
#define OFF_WL (OFF_HB + SZ_HB)
#define SZ_WL  (9 * GDIM * 4)              // 18432 (max of enc 9 rows / dec 8)
#define OFF_BL (OFF_WL + SZ_WL)
#define SZ_BL  (GDIM * 4)                  // 2048
#define OFF_ZS (OFF_BL + SZ_BL)
#define SZ_ZS  (GDIM * 4)                  // 2048
#define OFF_OW (OFF_ZS + SZ_ZS)
#define SZ_OW  (HDIM * 4)                  // 512
#define OFF_NE (OFF_OW + SZ_OW)
#define SZ_NE  (BT * 4)                    // 256
#define OFF_RS (OFF_NE + SZ_NE)
#define SZ_RS  (BT * 8 * 4)                // 2048
#define SMEM_TOTAL (OFF_RS + SZ_RS)        // 182016 bytes < 320KB/WGP

// --------------------------- small helpers ---------------------------------
__device__ __forceinline__ float sigf(float x) {
    return 1.0f / (1.0f + __expf(-x));
}
__device__ __forceinline__ float tanh_fast(float x) {
    // 2*sigmoid(2x)-1 : saturates cleanly at +-1, no inf/inf NaN path
    return 2.0f * sigf(2.0f * x) - 1.0f;
}

// A fragment: 16x32 (MxK) bf16, row-major source hbuf[m][k] (stride HS).
// Lane layout (ISA 7.12.2): lanes 0-15 -> M=lane, K chunks {k0..k0+7, k0+16..k0+23}
//                           lanes 16-31 -> M=lane-16, chunks start at k0+8.
__device__ __forceinline__ v16bf load_a_frag(const bf16_t* hbuf, int lane,
                                             int mbase, int kbase) {
    const int row = mbase + (lane & 15);
    const int k0  = kbase + ((lane >> 4) << 3);
    const bf16_t* p = hbuf + row * HS + k0;
    v16bf a;
    __builtin_memcpy(&a, p, 16);                          // K k0..k0+7
    __builtin_memcpy(((char*)&a) + 16, p + 16, 16);       // K k0+16..k0+23
    return a;
}

// B fragment: 32x16 (KxN) bf16, from transposed Ut[n][k] (stride UTS).
// Lane layout: lanes 0-15 -> N=lane, K chunks {k0..k0+7, k0+16..k0+23}; lanes 16-31 -> N=lane-16, chunks start at k0+8.
__device__ __forceinline__ v16bf load_b_frag(const bf16_t* Ut, int lane,
                                             int nbase, int kbase) {
    const int col = nbase + (lane & 15);
    const int k0  = kbase + ((lane >> 4) << 3);
    const bf16_t* p = Ut + col * UTS + k0;
    v16bf b;
    __builtin_memcpy(&b, p, 16);                          // K k0..k0+7
    __builtin_memcpy(((char*)&b) + 16, p + 16, 16);       // K k0+16..k0+23
    return b;
}

// Stage per-member weights into LDS: U transposed -> bf16, W/b kept f32.
__device__ __forceinline__ void load_weights(int tid, const float* __restrict__ U,
                                             const float* __restrict__ W, int wrows,
                                             const float* __restrict__ b,
                                             bf16_t* Ut, float* WL, float* bL) {
    for (int idx = tid; idx < HDIM * GDIM; idx += NTH) {
        const int k = idx >> 9;        // / 512  (coalesced global read)
        const int n = idx & 511;
        Ut[n * UTS + k] = (bf16_t)U[idx];
    }
    for (int idx = tid; idx < wrows * GDIM; idx += NTH) WL[idx] = W[idx];
    bL[tid] = b[tid];                  // NTH == GDIM
}

// One recurrent step. Template: encoder adds the y_prev * W_row0 term and
// offsets the feature rows of W by one.
template <bool IS_ENC>
__device__ __forceinline__ void lstm_step(
    int t, int tid, int lane, int mt, int htb, int bbase,
    const float* __restrict__ feats, const float* __restrict__ y_prev,
    const bf16_t* Ut, bf16_t* hbuf,
    const float* WL, const float* bL, float* zsL,
    float (&creg)[2][8])
{
    // ---- phase 1: batch-shared input term zs[n] = b[n] + feats(t) . W ----
    {
        const int n = tid;
        const float* fr = feats + t * DDIM;
        float s = bL[n];
        #pragma unroll
        for (int j = 0; j < DDIM; ++j)
            s += fr[j] * WL[(j + (IS_ENC ? 1 : 0)) * GDIM + n];
        zsL[n] = s;
    }

    // ---- phase 2: z += h(t-1) @ U via WMMA (bf16 -> f32) -----------------
    v8f acc[2][4];
    #pragma unroll
    for (int tp = 0; tp < 2; ++tp)
        #pragma unroll
        for (int g = 0; g < 4; ++g)
            #pragma unroll
            for (int q = 0; q < 8; ++q) acc[tp][g][q] = 0.0f;

    #pragma unroll
    for (int kk = 0; kk < 4; ++kk) {
        const v16bf a = load_a_frag(hbuf, lane, mt * 16, kk * 32);
        #pragma unroll
        for (int tp = 0; tp < 2; ++tp) {
            const int ht = htb + tp;
            #pragma unroll
            for (int g = 0; g < 4; ++g) {
                const v16bf b = load_b_frag(Ut, lane, (g * 8 + ht) * 16, kk * 32);
                acc[tp][g] = __builtin_amdgcn_wmma_f32_16x16x32_bf16(
                    false, a, false, b, (short)0, acc[tp][g], false, false);
            }
        }
    }
    __syncthreads();   // zs visible; all hbuf reads for this step retired

    // ---- phase 3: gates + state update (registers only) ------------------
    float yv[8];
    if (IS_ENC) {
        #pragma unroll
        for (int v = 0; v < 8; ++v) {
            const int m = mt * 16 + v + ((lane >> 4) << 3);
            yv[v] = y_prev[t * BATCH + bbase + m];
        }
    }
    #pragma unroll
    for (int tp = 0; tp < 2; ++tp) {
        const int hn = (htb + tp) * 16 + (lane & 15);
        const float zi0 = zsL[hn];
        const float zf0 = zsL[HDIM + hn];
        const float zg0 = zsL[2 * HDIM + hn];
        const float zo0 = zsL[3 * HDIM + hn];
        float wyi = 0.f, wyf = 0.f, wyg = 0.f, wyo = 0.f;
        if (IS_ENC) {                 // W row 0 (y coefficient)
            wyi = WL[hn]; wyf = WL[HDIM + hn];
            wyg = WL[2 * HDIM + hn]; wyo = WL[3 * HDIM + hn];
        }
        #pragma unroll
        for (int v = 0; v < 8; ++v) {
            const float y  = IS_ENC ? yv[v] : 0.0f;
            const float zi = acc[tp][0][v] + zi0 + y * wyi;
            const float zf = acc[tp][1][v] + zf0 + y * wyf;
            const float zg = acc[tp][2][v] + zg0 + y * wyg;
            const float zo = acc[tp][3][v] + zo0 + y * wyo;
            const float ig = sigf(zi);
            const float fg = sigf(zf);
            const float gg = tanh_fast(zg);
            const float og = sigf(zo);
            const float c  = fg * creg[tp][v] + ig * gg;
            creg[tp][v] = c;
            const float h = og * tanh_fast(c);
            const int m = mt * 16 + v + ((lane >> 4) << 3);
            hbuf[m * HS + hn] = (bf16_t)h;
        }
    }
    __syncthreads();   // h(t) visible for next step / output
}

__global__ void __launch_bounds__(NTH, 1)
lstm_ens_kernel(const float* __restrict__ feats,
                const float* __restrict__ y_prev,
                const int*   __restrict__ nid,
                const float* __restrict__ emb_table,
                const float* __restrict__ enc_W,
                const float* __restrict__ enc_U,
                const float* __restrict__ enc_b,
                const float* __restrict__ dec_W,
                const float* __restrict__ dec_U,
                const float* __restrict__ dec_b,
                const float* __restrict__ out_w,
                float* __restrict__ out)
{
    extern __shared__ char smem[];
    bf16_t* Ut   = (bf16_t*)(smem + OFF_UT);
    bf16_t* hbuf = (bf16_t*)(smem + OFF_HB);
    float*  WL   = (float*)(smem + OFF_WL);
    float*  bL   = (float*)(smem + OFF_BL);
    float*  zsL  = (float*)(smem + OFF_ZS);
    float*  owL  = (float*)(smem + OFF_OW);
    float*  nemb = (float*)(smem + OFF_NE);
    float*  rsum = (float*)(smem + OFF_RS);

    const int tid   = threadIdx.x;
    const int lane  = tid & 31;
    const int wv    = tid >> 5;
    const int mt    = wv & 3;            // M tile (rows mt*16..mt*16+15)
    const int htb   = (wv >> 2) << 1;    // base hn-tile (2 per wave)
    const int e     = blockIdx.y;
    const int bbase = blockIdx.x * BT;

    // normalized |embedding| weight for this member, per batch row
    if (tid < BT) {
        const int node = nid[bbase + tid];
        const float* er = emb_table + node * EDIM;
        float s = 0.f;
        #pragma unroll
        for (int j = 0; j < EDIM; ++j) s += fabsf(er[j]);
        nemb[tid] = fabsf(er[e]) / s;
    }
    // h0 = 0
    for (int i = tid; i < BT * HS; i += NTH) hbuf[i] = (bf16_t)0.0f;

    load_weights(tid, enc_U + e * HDIM * GDIM, enc_W + e * 9 * GDIM, 9,
                 enc_b + e * GDIM, Ut, WL, bL);
    __syncthreads();

    float creg[2][8];
    #pragma unroll
    for (int tp = 0; tp < 2; ++tp)
        #pragma unroll
        for (int v = 0; v < 8; ++v) creg[tp][v] = 0.0f;

    // -------------------------- encoder: 168 steps -------------------------
    #pragma unroll 1
    for (int t = 0; t < CONTL; ++t)
        lstm_step<true>(t, tid, lane, mt, htb, bbase, feats, y_prev,
                        Ut, hbuf, WL, bL, zsL, creg);

    // swap in decoder weights (all prior LDS reads retired at last barrier)
    load_weights(tid, dec_U + e * HDIM * GDIM, dec_W + e * DDIM * GDIM, DDIM,
                 dec_b + e * GDIM, Ut, WL, bL);
    if (tid < HDIM) owL[tid] = out_w[e * HDIM + tid];
    __syncthreads();

    // -------------------------- decoder: 24 steps --------------------------
    #pragma unroll 1
    for (int td = 0; td < HORIZ; ++td) {
        lstm_step<false>(CONTL + td, tid, lane, mt, htb, bbase, feats, y_prev,
                         Ut, hbuf, WL, bL, zsL, creg);

        // out[td, b] += nemb[b] * (h(td, b, :) . out_w)
        {
            const int r = tid >> 3;          // batch row 0..63
            const int sub = tid & 7;         // 8 partials per row
            const bf16_t* hr = hbuf + r * HS + sub * 16;
            const float*  ow = owL + sub * 16;
            float p = 0.f;
            #pragma unroll
            for (int q = 0; q < 16; ++q) p += (float)hr[q] * ow[q];
            rsum[r * 8 + sub] = p;
        }
        __syncthreads();
        if (tid < BT) {
            float s = 0.f;
            #pragma unroll
            for (int q = 0; q < 8; ++q) s += rsum[tid * 8 + q];
            atomicAdd(&out[td * BATCH + bbase + tid], s * nemb[tid]);
        }
    }
}

__global__ void zero_out_kernel(float* __restrict__ out, int n) {
    const int i = blockIdx.x * blockDim.x + threadIdx.x;
    if (i < n) out[i] = 0.0f;
}

extern "C" void kernel_launch(void* const* d_in, const int* in_sizes, int n_in,
                              void* d_out, int out_size, void* d_ws, size_t ws_size,
                              hipStream_t stream)
{
    const float* feats  = (const float*)d_in[0];
    const float* y_prev = (const float*)d_in[1];
    const int*   nid    = (const int*)  d_in[2];
    const float* emb    = (const float*)d_in[3];
    const float* encW   = (const float*)d_in[4];
    const float* encU   = (const float*)d_in[5];
    const float* encb   = (const float*)d_in[6];
    const float* decW   = (const float*)d_in[7];
    const float* decU   = (const float*)d_in[8];
    const float* decb   = (const float*)d_in[9];
    const float* outw   = (const float*)d_in[10];
    float* out = (float*)d_out;

    (void)hipFuncSetAttribute((const void*)lstm_ens_kernel,
                              hipFuncAttributeMaxDynamicSharedMemorySize,
                              SMEM_TOTAL);

    const int n = HORIZ * BATCH;
    zero_out_kernel<<<(n + 255) / 256, 256, 0, stream>>>(out, n);

    dim3 grid(BATCH / BT, EDIM);
    lstm_ens_kernel<<<grid, dim3(NTH), SMEM_TOTAL, stream>>>(
        feats, y_prev, nid, emb, encW, encU, encb, decW, decU, decb, outw, out);
}
